// SA_Layer_1133871366469
// MI455X (gfx1250) — matrix operand order repaired and checked by reading, not compiled, not run on hardware
//
#include <hip/hip_runtime.h>

typedef __attribute__((ext_vector_type(16))) __bf16 v16bf;
typedef __attribute__((ext_vector_type(8)))  __bf16 v8bf;
typedef __attribute__((ext_vector_type(8)))  float  v8f;
typedef __attribute__((ext_vector_type(4)))  int    v4i;

static constexpr int kB  = 16;
static constexpr int kC  = 256;
static constexpr int kC4 = 64;
static constexpr int kN  = 2048;
static constexpr float kEps = 1e-3f;

// ---------------- WMMA fragment helpers (wave32 layouts per CDNA5 ISA 7.12.2) ---
// All loaders issue exactly two 16-byte loads per lane (b128), no scalar gathers.

static __device__ inline v16bf cat16(v8bf lo, v8bf hi) {
  return __builtin_shufflevector(lo, hi, 0, 1, 2, 3, 4, 5, 6, 7,
                                 8, 9, 10, 11, 12, 13, 14, 15);
}

// A tile [16 x 32] bf16, row-major (M x K), leading dim ld (elements, mult of 8).
// Lane half h, row m: elems 0-7 = k(h*8 .. h*8+7), elems 8-15 = k(16+h*8 ..).
static __device__ inline v16bf frag_a_rm(const __bf16* p, int ld) {
  int lane = threadIdx.x & 31;
  int half = lane >> 4, m = lane & 15;
  const __bf16* q = p + m * ld + half * 8;
  v8bf lo = *(const v8bf*)(q);
  v8bf hi = *(const v8bf*)(q + 16);
  return cat16(lo, hi);
}

// B tile [32 x 16] (K x N) whose source is stored TRANSPOSED (N-major):
// element (k, n) lives at p[n*ld + k].  Lane: col n = lane&15, elems 0-7 =
// k(half*8 ..), elems 8-15 = k(16+half*8 ..): two b128 loads.
static __device__ inline v16bf frag_b_tr(const __bf16* p, int ld) {
  int lane = threadIdx.x & 31;
  int half = lane >> 4, n = lane & 15;
  const __bf16* q = p + n * ld + half * 8;
  v8bf lo = *(const v8bf*)(q);
  v8bf hi = *(const v8bf*)(q + 16);
  return cat16(lo, hi);
}

static __device__ inline v8f wmma_bf16(v16bf a, v16bf b, v8f c) {
  return __builtin_amdgcn_wmma_f32_16x16x32_bf16(
      /*neg_a=*/false, a, /*neg_b=*/false, b,
      /*c_mod=*/(short)0, c, /*reuse_a=*/false, /*reuse_b=*/false);
}

// ---------------- Kernel 0: weights fp32 -> bf16 -------------------------------

__global__ void k_wconv(const float* Wqk, const float* Wv, const float* Wt,
                        __bf16* Wqk_b, __bf16* Wv_b, __bf16* Wt_b) {
  int i = blockIdx.x * blockDim.x + threadIdx.x;
  if (i < kC4 * kC) Wqk_b[i] = (__bf16)Wqk[i];
  if (i < kC * kC) {
    Wv_b[i] = (__bf16)Wv[i];
    Wt_b[i] = (__bf16)Wt[i];
  }
}

// ---------------- Kernel 1: Kt = (Wqk*x)^T, V = Wv*x + bv  (bf16) --------------
// grid = kB * (kN/128), block = 256 (8 waves).
// x slab staged TRANSPOSED in LDS: xs_t[n][c] so B-fragments are contiguous.
// K stored transposed Kt[b][n][64] so the Gram matrix E = K^T K can read both
// A and B fragments from it with contiguous b128 loads.

__global__ void k_proj(const float* x, const __bf16* Wqk_b, const __bf16* Wv_b,
                       const float* bv, __bf16* Kt, __bf16* Vb) {
  __shared__ __align__(16) __bf16 xs_t[128][kC];  // 64 KB
  int b  = blockIdx.x >> 4;
  int n0 = (blockIdx.x & 15) * 128;
  const float* xb = x + (size_t)b * kC * kN;

  for (int j = threadIdx.x; j < kC * 128; j += 256) {
    int c = j >> 7, n = j & 127;                 // consecutive threads: stride-1 n
    xs_t[n][c] = (__bf16)xb[c * kN + n0 + n];
  }
  __syncthreads();

  int w = threadIdx.x >> 5;
  int lane = threadIdx.x & 31, half = lane >> 4, col = lane & 15;
  __bf16* Ktb = Kt + (size_t)b * kN * kC4;
  __bf16* Vbb = Vb + (size_t)b * kC * kN;

  // 32 K-tiles then 128 V-tiles, round-robin over the 8 waves.
  for (int t = w; t < 160; t += 8) {
    bool isK = (t < 32);
    int idx = isK ? t : t - 32;
    int tr = idx >> 3, tc = idx & 7;
    const __bf16* W = isK ? Wqk_b : Wv_b;
    v8f acc = {};
#pragma unroll
    for (int kk = 0; kk < kC; kk += 32) {
      v16bf a = frag_a_rm(W + (tr * 16) * kC + kk, kC);
      v16bf bb = frag_b_tr(&xs_t[tc * 16][kk], kC);
      acc = wmma_bf16(a, bb, acc);
    }
#pragma unroll
    for (int r = 0; r < 8; ++r) {
      int row = tr * 16 + half * 8 + r;          // output channel
      int n = tc * 16 + col;
      if (isK) Ktb[(n0 + n) * kC4 + row] = (__bf16)acc[r];
      else     Vbb[row * kN + n0 + n] = (__bf16)(acc[r] + bv[row]);
    }
  }
}

// ---------------- Kernel 2: row softmax stats over E = K^T K -------------------
// grid = kB * (kN/16); block = 256 (8 waves).  Stores rowmax and 1/rowsum.

__global__ void k_rowstats(const __bf16* Kt, float* rowmax, float* rowinv) {
  __shared__ __align__(16) float et[8][16][16];
  __shared__ float mp[8][16], sp[8][16];
  int b  = blockIdx.x >> 7;
  int n0 = (blockIdx.x & 127) * 16;
  const __bf16* Ktb = Kt + (size_t)b * kN * kC4;

  int w = threadIdx.x >> 5;
  int lane = threadIdx.x & 31, half = lane >> 4, col = lane & 15;
  float m_run = -3.0e38f, s_run = 0.0f;

  for (int tc = w; tc < kN / 16; tc += 8) {
    if (tc + 8 < kN / 16)
      __builtin_prefetch(Ktb + (tc + 8) * 16 * kC4, 0, 1);
    v8f acc = {};
#pragma unroll
    for (int kk = 0; kk < kC4; kk += 32) {
      v16bf a = frag_a_rm(Ktb + n0 * kC4 + kk, kC4);        // E rows (n)
      v16bf bb = frag_b_tr(Ktb + (tc * 16) * kC4 + kk, kC4); // E cols (m)
      acc = wmma_bf16(a, bb, acc);
    }
#pragma unroll
    for (int r = 0; r < 8; ++r) et[w][half * 8 + r][col] = acc[r];
    asm volatile("s_wait_dscnt 0" ::: "memory");  // intra-wave LDS RAW fence
    if (lane < 16) {
      float tmax = -3.0e38f;
#pragma unroll
      for (int j = 0; j < 16; ++j) tmax = fmaxf(tmax, et[w][lane][j]);
      float m_new = fmaxf(m_run, tmax);
      float s = s_run * __expf(m_run - m_new);
#pragma unroll
      for (int j = 0; j < 16; ++j) s += __expf(et[w][lane][j] - m_new);
      m_run = m_new; s_run = s;
    }
  }
  if (lane < 16) { mp[w][lane] = m_run; sp[w][lane] = s_run; }
  __syncthreads();
  if (threadIdx.x < 16) {
    int r = threadIdx.x;
    float M = -3.0e38f;
#pragma unroll
    for (int j = 0; j < 8; ++j) M = fmaxf(M, mp[j][r]);
    float S = 0.0f;
#pragma unroll
    for (int j = 0; j < 8; ++j) S += sp[j][r] * __expf(mp[j][r] - M);
    rowmax[b * kN + n0 + r] = M;
    rowinv[b * kN + n0 + r] = 1.0f / S;
  }
}

// ---------------- Kernel 3: colinv = 1/(1e-9 + sum_n attn[n,m]) ----------------
// grid = kB * (kN/16); block = 256 (8 waves).

__global__ void k_colsum(const __bf16* Kt, const float* rowmax,
                         const float* rowinv, float* colinv) {
  __shared__ __align__(16) float et[8][16][16];
  __shared__ float pp[8][16];
  int b  = blockIdx.x >> 7;
  int m0 = (blockIdx.x & 127) * 16;
  const __bf16* Ktb = Kt + (size_t)b * kN * kC4;
  const float* rm = rowmax + b * kN;
  const float* ri = rowinv + b * kN;

  int w = threadIdx.x >> 5;
  int lane = threadIdx.x & 31, half = lane >> 4, col = lane & 15;
  float part = 0.0f;

  for (int tn = w; tn < kN / 16; tn += 8) {
    if (tn + 8 < kN / 16)
      __builtin_prefetch(Ktb + (tn + 8) * 16 * kC4, 0, 1);
    v8f acc = {};
#pragma unroll
    for (int kk = 0; kk < kC4; kk += 32) {
      v16bf a = frag_a_rm(Ktb + (tn * 16) * kC4 + kk, kC4);
      v16bf bb = frag_b_tr(Ktb + m0 * kC4 + kk, kC4);
      acc = wmma_bf16(a, bb, acc);
    }
#pragma unroll
    for (int r = 0; r < 8; ++r) et[w][half * 8 + r][col] = acc[r];
    asm volatile("s_wait_dscnt 0" ::: "memory");
    if (lane < 16) {
#pragma unroll
      for (int r = 0; r < 16; ++r) {
        int n = tn * 16 + r;
        part += __expf(et[w][r][lane] - rm[n]) * ri[n];
      }
    }
  }
  if (lane < 16) pp[w][lane] = part;
  __syncthreads();
  if (threadIdx.x < 16) {
    float S = 0.0f;
#pragma unroll
    for (int j = 0; j < 8; ++j) S += pp[j][threadIdx.x];
    colinv[b * kN + m0 + threadIdx.x] = 1.0f / (1e-9f + S);
  }
}

// ---------------- Kernel 4: XR = V * attn (flash-style), emit (x - XR)^T bf16 --
// grid = kB * (kN/32) * (kC/128); block = 256 (8 waves).
// Per 32-wide n-step: waves 0-3 recompute the 32x32 E tile, normalize and store
// the attn tile TRANSPOSED (abuf_t[m][n]) in LDS; all 8 waves then run the
// 128x32 XR accumulation.  Output (x - XR) is written transposed so the final
// GEMM can stage its slab as one contiguous block.

__global__ void k_xr(const float* x, const __bf16* Kt, const __bf16* Vb,
                     const float* rowmax, const float* rowinv,
                     const float* colinv, __bf16* XmRt) {
  __shared__ __align__(16) __bf16 abuf_t[32][32];  // [m][n]
  int b   = blockIdx.x >> 7;
  int rem = blockIdx.x & 127;
  int m0  = (rem >> 1) * 32;
  int c0  = (rem & 1) * 128;
  const __bf16* Ktb = Kt + (size_t)b * kN * kC4;
  const __bf16* Vbb = Vb + (size_t)b * kC * kN;
  const float* rm = rowmax + b * kN;
  const float* ri = rowinv + b * kN;
  const float* ci = colinv + b * kN;

  int w = threadIdx.x >> 5;
  int lane = threadIdx.x & 31, half = lane >> 4, col = lane & 15;

  v8f acc0 = {}, acc1 = {};
  for (int nn = 0; nn < kN; nn += 32) {
    if (nn + 32 < kN)
      __builtin_prefetch(Vbb + (c0 + w * 16) * kN + nn + 32, 0, 1);
    if (w < 4) {  // E subtile (er,ec) of the 32x32 block
      int er = w >> 1, ec = w & 1;
      v8f e = {};
#pragma unroll
      for (int kk = 0; kk < kC4; kk += 32) {
        v16bf a = frag_a_rm(Ktb + (nn + er * 16) * kC4 + kk, kC4);
        v16bf bb = frag_b_tr(Ktb + (m0 + ec * 16) * kC4 + kk, kC4);
        e = wmma_bf16(a, bb, e);
      }
#pragma unroll
      for (int r = 0; r < 8; ++r) {
        int nl = er * 16 + half * 8 + r;
        int ml = ec * 16 + col;
        int n = nn + nl, m = m0 + ml;
        float av = __expf(e[r] - rm[n]) * ri[n] * ci[m];
        abuf_t[ml][nl] = (__bf16)av;
      }
    }
    __syncthreads();
    // XR accumulation: wave w owns rows c0 + w*16 .. +15, both 16-col halves.
    v16bf va = frag_a_rm(Vbb + (c0 + w * 16) * kN + nn, kN);
    acc0 = wmma_bf16(va, frag_b_tr(&abuf_t[0][0], 32), acc0);
    acc1 = wmma_bf16(va, frag_b_tr(&abuf_t[16][0], 32), acc1);
    __syncthreads();
  }

  const float* xb = x + (size_t)b * kC * kN;
  __bf16* ot = XmRt + (size_t)b * kN * kC;
#pragma unroll
  for (int r = 0; r < 8; ++r) {
    int c = c0 + w * 16 + half * 8 + r;
    int m = m0 + col;
    ot[m * kC + c]        = (__bf16)(xb[c * kN + m]      - acc0[r]);
    ot[(m + 16) * kC + c] = (__bf16)(xb[c * kN + m + 16] - acc1[r]);
  }
}

// ---------------- Kernel 5: out = x + relu(BN(Wt*(x-XR) + bt)) -----------------
// grid = kB * (kN/128); block = 256 (8 waves).
// (x-XR)^T slab [n][c] is a CONTIGUOUS 64 KB global block -> async copy to LDS.

__global__ void k_final(const float* x, const __bf16* XmRt, const __bf16* Wt_b,
                        const float* bt, const float* gamma, const float* beta,
                        const float* mean, const float* var, float* out) {
  __shared__ __align__(16) __bf16 st[128][kC];  // 64 KB, [n][c]
  int b  = blockIdx.x >> 4;
  int n0 = (blockIdx.x & 15) * 128;
  const __bf16* xmt = XmRt + (size_t)b * kN * kC + (size_t)n0 * kC;

#if __has_builtin(__builtin_amdgcn_global_load_async_to_lds_b128)
  {
    typedef __attribute__((address_space(1))) v4i gv4i;  // global
    typedef __attribute__((address_space(3))) v4i lv4i;  // LDS
    gv4i* g = (gv4i*)xmt;
    lv4i* l = (lv4i*)&st[0][0];
    for (int j = threadIdx.x; j < (128 * kC * 2) / 16; j += 256)
      __builtin_amdgcn_global_load_async_to_lds_b128(g + j, l + j, 0, 0);
    asm volatile("s_wait_asynccnt 0" ::: "memory");
  }
#else
  for (int j = threadIdx.x; j < 128 * kC; j += 256) {
    int n = j >> 8, c = j & 255;
    st[n][c] = xmt[n * kC + c];
  }
#endif
  __syncthreads();

  int w = threadIdx.x >> 5;
  int lane = threadIdx.x & 31, half = lane >> 4, col = lane & 15;
  const float* xb = x + (size_t)b * kC * kN;
  float* ob = out + (size_t)b * kC * kN;

  for (int t = w; t < 128; t += 8) {
    int tr = t >> 3, tc = t & 7;
    v8f acc = {};
#pragma unroll
    for (int kk = 0; kk < kC; kk += 32) {
      v16bf a = frag_a_rm(Wt_b + (tr * 16) * kC + kk, kC);
      v16bf bb = frag_b_tr(&st[tc * 16][kk], kC);
      acc = wmma_bf16(a, bb, acc);
    }
#pragma unroll
    for (int r = 0; r < 8; ++r) {
      int c = tr * 16 + half * 8 + r;
      int n = n0 + tc * 16 + col;
      float y = acc[r] + bt[c];
      y = (y - mean[c]) * rsqrtf(var[c] + kEps) * gamma[c] + beta[c];
      ob[c * kN + n] = xb[c * kN + n] + fmaxf(y, 0.0f);
    }
  }
}

// ---------------- Launch -------------------------------------------------------

extern "C" void kernel_launch(void* const* d_in, const int* in_sizes, int n_in,
                              void* d_out, int out_size, void* d_ws, size_t ws_size,
                              hipStream_t stream) {
  const float* x     = (const float*)d_in[0];
  const float* Wqk   = (const float*)d_in[1];
  const float* Wv    = (const float*)d_in[2];
  const float* bv    = (const float*)d_in[3];
  const float* Wt    = (const float*)d_in[4];
  const float* bt    = (const float*)d_in[5];
  const float* gamma = (const float*)d_in[6];
  const float* beta  = (const float*)d_in[7];
  const float* mean  = (const float*)d_in[8];
  const float* var   = (const float*)d_in[9];
  float* out = (float*)d_out;

  char* w = (char*)d_ws;
  size_t off = 0;
  auto alloc = [&](size_t bytes) {
    void* p = w + off;
    off = (off + bytes + 255) & ~(size_t)255;
    return p;
  };
  __bf16* Wqk_b  = (__bf16*)alloc((size_t)kC4 * kC * 2);
  __bf16* Wv_b   = (__bf16*)alloc((size_t)kC * kC * 2);
  __bf16* Wt_b   = (__bf16*)alloc((size_t)kC * kC * 2);
  __bf16* Kt     = (__bf16*)alloc((size_t)kB * kN * kC4 * 2);   // [b][n][64]
  __bf16* Vb     = (__bf16*)alloc((size_t)kB * kC * kN * 2);    // [b][c][n]
  float*  rowmax = (float*)alloc((size_t)kB * kN * 4);
  float*  rowinv = (float*)alloc((size_t)kB * kN * 4);
  float*  colinv = (float*)alloc((size_t)kB * kN * 4);
  __bf16* XmRt   = (__bf16*)alloc((size_t)kB * kN * kC * 2);    // [b][n][c]

  k_wconv<<<kC * kC / 256, 256, 0, stream>>>(Wqk, Wv, Wt, Wqk_b, Wv_b, Wt_b);
  k_proj<<<kB * (kN / 128), 256, 0, stream>>>(x, Wqk_b, Wv_b, bv, Kt, Vb);
  k_rowstats<<<kB * (kN / 16), 256, 0, stream>>>(Kt, rowmax, rowinv);
  k_colsum<<<kB * (kN / 16), 256, 0, stream>>>(Kt, rowmax, rowinv, colinv);
  k_xr<<<kB * (kN / 32) * (kC / 128), 256, 0, stream>>>(x, Kt, Vb, rowmax,
                                                        rowinv, colinv, XmRt);
  k_final<<<kB * (kN / 128), 256, 0, stream>>>(x, XmRt, Wt_b, bt, gamma, beta,
                                               mean, var, out);
}
